// LocMoEPlusLayer_48593259987196
// MI455X (gfx1250) — matrix operand order repaired and checked by reading, not compiled, not run on hardware
//
#include <hip/hip_runtime.h>

// ---------------- problem constants ----------------
#define BB 4
#define SS 2048
#define DD 1024
#define EE 8
#define FF 4096
#define MINCAP 4

typedef __attribute__((ext_vector_type(16))) __bf16 v16bf;
typedef __attribute__((ext_vector_type(8)))  __bf16 v8bf;
typedef __attribute__((ext_vector_type(8)))  float  v8f;

// ---------------- helpers ----------------
static __device__ __forceinline__ unsigned short f32_bf16_rne(float f) {
  unsigned u = __float_as_uint(f);
  return (unsigned short)((u + 0x7fffu + ((u >> 16) & 1u)) >> 16);
}
static __device__ __forceinline__ float bf16_f32(unsigned short s) {
  return __uint_as_float(((unsigned)s) << 16);
}
static __device__ __forceinline__ void split_bf(float v, unsigned short& hi, unsigned short& lo) {
  hi = f32_bf16_rne(v);
  lo = f32_bf16_rne(v - bf16_f32(hi));
}
static __device__ __forceinline__ float gelu_f(float x) {
  return 0.5f * x * (1.0f + erff(x * 0.70710678118654752f));
}
static __device__ __forceinline__ float wredf(float v) {
  #pragma unroll
  for (int o = 16; o > 0; o >>= 1) v += __shfl_xor(v, o, 32);
  return v;
}
static __device__ __forceinline__ int wredi(int v) {
  #pragma unroll
  for (int o = 16; o > 0; o >>= 1) v += __shfl_xor(v, o, 32);
  return v;
}
static __device__ __forceinline__ v8f wmma_bf16(v16bf a, v16bf b, v8f c) {
  return __builtin_amdgcn_wmma_f32_16x16x32_bf16(false, a, false, b, (short)0, c, false, false);
}
// A operand: per-lane 16 contiguous bf16 = K{o..o+15}  (two 16B LDS loads)
static __device__ __forceinline__ v16bf ldsA(const unsigned short* p) {
  v8bf a = *(const v8bf*)p;
  v8bf b = *(const v8bf*)(p + 8);
  v16bf r;
  #pragma unroll
  for (int i = 0; i < 8; i++) { r[i] = a[i]; r[i + 8] = b[i]; }
  return r;
}
// B operand: per-lane 16 contiguous bf16 (two 16B LDS loads)
static __device__ __forceinline__ v16bf ldsB(const unsigned short* p) {
  v8bf a = *(const v8bf*)p;
  v8bf b = *(const v8bf*)(p + 8);
  v16bf r;
  #pragma unroll
  for (int i = 0; i < 8; i++) { r[i] = a[i]; r[i + 8] = b[i]; }
  return r;
}

// ---------------- kernel 0: zero counts ----------------
__global__ void k_init(int* counts) {
  if (threadIdx.x < BB * EE) counts[threadIdx.x] = 0;
}

// ---------------- kernel 1: affinity (cosine) + partial sums ----------------
__global__ __launch_bounds__(256) void k_aff(const float* __restrict__ x,
                                             const float* __restrict__ waff,
                                             float* __restrict__ aff,
                                             float* __restrict__ partial) {
  __shared__ float wl[EE * DD];     // 32 KB
  __shared__ float wsum[8];
  for (int i = threadIdx.x; i < EE * DD; i += 256) wl[i] = waff[i];
  __syncthreads();
  const int lane = threadIdx.x & 31, wave = threadIdx.x >> 5;
  const int gt = blockIdx.x * 8 + wave;           // flattened token id 0..B*S-1
  float xx = 0.f, num[EE], ww[EE];
  #pragma unroll
  for (int e = 0; e < EE; e++) { num[e] = 0.f; ww[e] = 0.f; }
  for (int d = lane; d < DD; d += 32) {
    float xv = x[(size_t)gt * DD + d];
    xx += xv * xv;
    #pragma unroll
    for (int e = 0; e < EE; e++) {
      float wv = wl[e * DD + d];
      num[e] += xv * wv;
      ww[e]  += wv * wv;
    }
  }
  xx = wredf(xx);
  float s = 0.f;
  #pragma unroll
  for (int e = 0; e < EE; e++) {
    float nu = wredf(num[e]);
    float w2 = wredf(ww[e]);
    float a = nu / (sqrtf(xx) * sqrtf(w2) + 1e-9f);
    if (lane == 0) aff[(size_t)gt * EE + e] = a;
    s += a;
  }
  if (lane == 0) wsum[wave] = s;
  __syncthreads();
  if (threadIdx.x == 0) {
    float t = 0.f;
    for (int w = 0; w < 8; w++) t += wsum[w];   // fixed order -> deterministic
    partial[blockIdx.x] = t;
  }
}

// ---------------- kernel 2: adaptive capacity ----------------
__global__ void k_cap(const float* __restrict__ partial, const float* __restrict__ thr,
                      int* __restrict__ capv) {
  const int lane = threadIdx.x;
  float s = 0.f;
  for (int i = lane; i < 1024; i += 32) s += partial[i];
  s = wredf(s);
  if (lane == 0) {
    float mean = s / (float)(BB * SS * EE);
    float sig = 1.f / (1.f + expf(-(mean - thr[0])));
    int cap = (int)floorf((float)SS * sig);
    cap = cap < MINCAP ? MINCAP : cap;
    cap = cap > SS ? SS : cap;
    capv[0] = cap;
  }
}

// ---------------- kernel 3: argmax + rank + compaction ----------------
__global__ __launch_bounds__(256) void k_disp(const float* __restrict__ aff,
                                              const int* __restrict__ capv,
                                              int* __restrict__ counts,
                                              int* __restrict__ lists) {
  const int lane = threadIdx.x & 31, wave = threadIdx.x >> 5;
  const int gt = blockIdx.x * 8 + wave;
  const int b = gt >> 11;              // / S
  const int t = gt & (SS - 1);
  int estar = 0; float am = 0.f;
  if (lane == 0) {
    am = aff[(size_t)gt * EE];
    #pragma unroll
    for (int e = 1; e < EE; e++) {
      float v = aff[(size_t)gt * EE + e];
      if (v > am) { am = v; estar = e; }          // first-max: matches argmax
    }
  }
  estar = __shfl(estar, 0, 32);
  am    = __shfl(am, 0, 32);
  const float* col = aff + (size_t)b * SS * EE + estar;
  int c = 0;
  for (int tp = lane; tp < SS; tp += 32) {
    float v = col[(size_t)tp * EE];
    c += (v > am) || (v == am && tp < t);         // stable-sort rank
  }
  c = wredi(c);
  if (lane == 0 && c < capv[0]) {
    int pos = atomicAdd(&counts[b * EE + estar], 1);
    lists[(size_t)(b * EE + estar) * SS + pos] = t;
  }
}

// ---------------- kernel 4a: gelu(b1) ----------------
__global__ void k_hb(const float* __restrict__ b1, float* __restrict__ hb) {
  int i = blockIdx.x * 256 + threadIdx.x;
  if (i < EE * FF) hb[i] = gelu_f(b1[i]);
}

// ---------------- kernel 4b: base[d] = sum_e (gelu(b1e) @ W2e + b2e)[d] ----------------
__global__ void k_base(const float* __restrict__ hb, const float* __restrict__ W2,
                       const float* __restrict__ b2, float* __restrict__ basev) {
  int d = blockIdx.x * 256 + threadIdx.x;          // 1024 total
  float s = 0.f;
  for (int e = 0; e < EE; e++) {
    s += b2[e * DD + d];
    const float* w2p = W2 + (size_t)e * FF * DD + d;
    const float* hbp = hb + e * FF;
    for (int f = 0; f < FF; f++) s += hbp[f] * w2p[(size_t)f * DD];
  }
  basev[d] = s;
}

// ---------------- kernel 5: out = base (broadcast) ----------------
__global__ void k_outinit(const float* __restrict__ basev, float* __restrict__ out) {
  size_t i = (size_t)blockIdx.x * 256 + threadIdx.x;
  if (i < (size_t)BB * SS * DD) out[i] = basev[i & (DD - 1)];
}

// ---------------- kernel 6: fused expert FFN (split-bf16 WMMA) ----------------
// block: 256 threads (8 waves). 32-token tile per block, each wave owns 128 output cols.
__global__ __launch_bounds__(256) void k_ffn(const float* __restrict__ x,
                                             const int* __restrict__ lists,
                                             const int* __restrict__ counts,
                                             const float* __restrict__ W1,
                                             const float* __restrict__ b1,
                                             const float* __restrict__ hb,
                                             const float* __restrict__ W2,
                                             const float* __restrict__ basev,
                                             float* __restrict__ out) {
  constexpr int MT = 32;           // token tile
  constexpr int SXA = 1032;        // X row stride (bf16 elems), padded
  constexpr int SH  = 136;         // H row stride
  constexpr int SW  = 40;          // W2-transposed row stride (K=32 + pad)
  constexpr int SW1 = 136;         // W1-transposed row stride (K=128 + pad)

  extern __shared__ unsigned short sm[];
  unsigned short* xh = sm;                         // [MT][SXA]
  unsigned short* xl = xh + MT * SXA;
  unsigned short* hh = xl + MT * SXA;              // [MT][SH]
  unsigned short* hl = hh + MT * SH;
  unsigned short* wh = hl + MT * SH;               // staging, hi part
  unsigned short* wl = wh + 1024 * SW;             // staging, lo part
  __shared__ int tmap[MT];

  const int be = blockIdx.y;
  const int b = be >> 3, e = be & 7;
  const int n = counts[be];
  const int t0 = blockIdx.x * MT;
  if (t0 >= n) return;
  const int tid = threadIdx.x, lane = tid & 31, wave = tid >> 5;
  const int tc = min(MT, n - t0);
  if (tid < MT) tmap[tid] = (tid < tc) ? lists[(size_t)be * SS + t0 + tid] : -1;
  __syncthreads();

  // load + split the X tile into LDS (hi/lo bf16)
  for (int i = tid; i < MT * DD; i += 256) {
    int r = i >> 10, d = i & (DD - 1);
    int tok = tmap[r];
    float v = (tok >= 0) ? x[((size_t)b * SS + tok) * DD + d] : 0.f;
    unsigned short hi, lo; split_bf(v, hi, lo);
    xh[r * SXA + d] = hi;
    xl[r * SXA + d] = lo;
  }

  const int col = lane & 15, kh = lane >> 4;
  v8f acc2[2][8];
  const v8f vzero = {0.f, 0.f, 0.f, 0.f, 0.f, 0.f, 0.f, 0.f};
  #pragma unroll
  for (int mt = 0; mt < 2; mt++)
    #pragma unroll
    for (int nt = 0; nt < 8; nt++) acc2[mt][nt] = vzero;

  for (int fc = 0; fc < FF; fc += 128) {
    // ---------- GEMM1: H[32,128] = X[32,1024] @ W1[1024, fc:fc+128] ----------
    v8f acc1[2] = {vzero, vzero};
    for (int kb = 0; kb < DD; kb += 128) {
      __syncthreads();
      for (int i = tid; i < 128 * 128; i += 256) {   // stage W1 chunk, transposed
        int k = i >> 7, f = i & 127;
        float v = W1[((size_t)e * DD + kb + k) * FF + fc + f];
        unsigned short hi, lo; split_bf(v, hi, lo);
        wh[f * SW1 + k] = hi;
        wl[f * SW1 + k] = lo;
      }
      __syncthreads();
      #pragma unroll
      for (int ks = 0; ks < 4; ks++) {
        const int bo = (wave * 16 + col) * SW1 + ks * 32 + kh * 16;
        v16bf bhv = ldsB(wh + bo);
        v16bf blv = ldsB(wl + bo);
        #pragma unroll
        for (int mt = 0; mt < 2; mt++) {
          const int r = mt * 16 + col;
          const int o = kb + ks * 32 + kh * 16;
          v16bf ahv = ldsA(xh + r * SXA + o);
          v16bf alv = ldsA(xl + r * SXA + o);
          acc1[mt] = wmma_bf16(ahv, bhv, acc1[mt]);
          acc1[mt] = wmma_bf16(ahv, blv, acc1[mt]);
          acc1[mt] = wmma_bf16(alv, bhv, acc1[mt]);
        }
      }
    }
    // activation: h = gelu(acc + b1) - gelu(b1); split + store to LDS
    const int fcol = fc + wave * 16 + col;
    const float b1v = b1[e * FF + fcol];
    const float hbv = hb[e * FF + fcol];
    #pragma unroll
    for (int mt = 0; mt < 2; mt++) {
      #pragma unroll
      for (int v = 0; v < 8; v++) {
        int r = mt * 16 + (kh ? v + 8 : v);
        float hv = gelu_f(acc1[mt][v] + b1v) - hbv;
        unsigned short hi, lo; split_bf(hv, hi, lo);
        hh[r * SH + wave * 16 + col] = hi;
        hl[r * SH + wave * 16 + col] = lo;
      }
    }
    // ---------- GEMM2: acc2 += H[32,128] @ W2[fc:fc+128, 0:1024] ----------
    for (int ks = 0; ks < 4; ks++) {
      __syncthreads();
      for (int i = tid; i < 32 * 1024; i += 256) {   // stage W2 chunk, transposed
        int k = i >> 10, d = i & 1023;
        float v = W2[((size_t)e * FF + fc + ks * 32 + k) * DD + d];
        unsigned short hi, lo; split_bf(v, hi, lo);
        wh[d * SW + k] = hi;
        wl[d * SW + k] = lo;
      }
      __syncthreads();
      v16bf ah2[2], al2[2];
      #pragma unroll
      for (int mt = 0; mt < 2; mt++) {
        const int r = mt * 16 + col;
        const int o = ks * 32 + kh * 16;
        ah2[mt] = ldsA(hh + r * SH + o);
        al2[mt] = ldsA(hl + r * SH + o);
      }
      #pragma unroll
      for (int nt = 0; nt < 8; nt++) {
        const int drow = wave * 128 + nt * 16 + col;
        v16bf bhv = ldsB(wh + drow * SW + kh * 16);
        v16bf blv = ldsB(wl + drow * SW + kh * 16);
        #pragma unroll
        for (int mt = 0; mt < 2; mt++) {
          acc2[mt][nt] = wmma_bf16(ah2[mt], bhv, acc2[mt][nt]);
          acc2[mt][nt] = wmma_bf16(ah2[mt], blv, acc2[mt][nt]);
          acc2[mt][nt] = wmma_bf16(al2[mt], bhv, acc2[mt][nt]);
        }
      }
    }
  }
  // epilogue: out[token, d] = base[d] + acc
  #pragma unroll
  for (int mt = 0; mt < 2; mt++) {
    #pragma unroll
    for (int v = 0; v < 8; v++) {
      int i = mt * 16 + (kh ? v + 8 : v);
      int tok = tmap[i];
      if (tok >= 0) {
        #pragma unroll
        for (int nt = 0; nt < 8; nt++) {
          int d = wave * 128 + nt * 16 + col;
          out[((size_t)b * SS + tok) * DD + d] = basev[d] + acc2[mt][nt][v];
        }
      }
    }
  }
}

// ---------------- host launcher ----------------
extern "C" void kernel_launch(void* const* d_in, const int* in_sizes, int n_in,
                              void* d_out, int out_size, void* d_ws, size_t ws_size,
                              hipStream_t stream) {
  const float* x    = (const float*)d_in[0];
  const float* waff = (const float*)d_in[1];
  const float* W1   = (const float*)d_in[2];
  const float* b1   = (const float*)d_in[3];
  const float* W2   = (const float*)d_in[4];
  const float* b2   = (const float*)d_in[5];
  const float* thr  = (const float*)d_in[6];
  float* out = (float*)d_out;

  float* ws      = (float*)d_ws;
  float* aff     = ws;                          // B*S*E
  float* partial = aff + BB * SS * EE;          // 1024
  int*   capv    = (int*)(partial + 1024);      // 1
  int*   counts  = capv + 1;                    // B*E
  int*   lists   = counts + BB * EE;            // B*E*S
  float* hb      = (float*)(lists + BB * EE * SS);  // E*F
  float* basev   = hb + EE * FF;                // D

  k_init<<<1, 64, 0, stream>>>(counts);
  k_aff<<<BB * SS / 8, 256, 0, stream>>>(x, waff, aff, partial);
  k_cap<<<1, 32, 0, stream>>>(partial, thr, capv);
  k_disp<<<BB * SS / 8, 256, 0, stream>>>(aff, capv, counts, lists);
  k_hb<<<(EE * FF + 255) / 256, 256, 0, stream>>>(b1, hb);
  k_base<<<DD / 256, 256, 0, stream>>>(hb, W2, b2, basev);
  k_outinit<<<(int)(((size_t)BB * SS * DD + 255) / 256), 256, 0, stream>>>(basev, out);

  constexpr int MT = 32, SXA = 1032, SH = 136, SW = 40;
  const size_t smem = (size_t)(2 * MT * SXA + 2 * MT * SH + 2 * 1024 * SW) * sizeof(unsigned short);
  dim3 grid(SS / MT, BB * EE);
  k_ffn<<<grid, 256, smem, stream>>>(x, lists, counts, W1, b1, hb, W2, basev, out);
}
